// AnimaControlSelfAttn_32323923869985
// MI455X (gfx1250) — matrix operand order repaired and checked by reading, not compiled, not run on hardware
//
#include <hip/hip_runtime.h>
#include <hip/hip_bf16.h>

typedef __bf16 bf16;
typedef __attribute__((ext_vector_type(16))) __bf16 bf16x16;
typedef __attribute__((ext_vector_type(8)))  float  f32x8;

union U16 { unsigned u[8]; bf16x16 v; };

#define L_NOISE 2048
#define L_COND  1024
#define LFULL   3072
#define DIM     2048
#define NH      16
#define HD      128
#define RANK_   128
#define EPSV    1e-6f

// A-layout fragment: pairs at +0,2,4,6 and +16,18,20,22 from p
__device__ __forceinline__ bf16x16 load_frag_a(const bf16* p) {
    U16 u;
    u.u[0] = *(const unsigned*)(p + 0);  u.u[1] = *(const unsigned*)(p + 2);
    u.u[2] = *(const unsigned*)(p + 4);  u.u[3] = *(const unsigned*)(p + 6);
    u.u[4] = *(const unsigned*)(p + 16); u.u[5] = *(const unsigned*)(p + 18);
    u.u[6] = *(const unsigned*)(p + 20); u.u[7] = *(const unsigned*)(p + 22);
    return u.v;
}
// B-layout fragment: pairs at +0,2,4,6 and +16,18,20,22 from p
__device__ __forceinline__ bf16x16 load_frag_b(const bf16* p) {
    U16 u;
#pragma unroll
    for (int j = 0; j < 8; ++j) u.u[j] = *(const unsigned*)(p + 2 * j + (j >= 4 ? 8 : 0));
    return u.v;
}

// ---------------------------------------------------------------- cast f32->bf16 (pairs)
__global__ void cast_f32_bf16x2(const float2* __restrict__ s, unsigned* __restrict__ d,
                                long long n2) {
    long long i = (long long)blockIdx.x * blockDim.x + threadIdx.x;
    long long stride = (long long)gridDim.x * blockDim.x;
    for (; i < n2; i += stride) {
        float2 v = s[i];
        union { bf16 h[2]; unsigned u; } p;
        p.h[0] = (bf16)v.x; p.h[1] = (bf16)v.y;
        d[i] = p.u;
    }
}

// ---------------------------------------------------------------- cast + transpose
// dst[c*R + r] = (bf16)src[r*C + c]; R, C multiples of 32. 256 threads, 32x32 tiles.
__global__ __launch_bounds__(256)
void cast_transpose_bf16(const float* __restrict__ src, bf16* __restrict__ dst,
                         int R, int C) {
    __shared__ float tile[32][33];
    const int tx = threadIdx.x & 31, ty = threadIdx.x >> 5;  // ty 0..7
    const int r0 = blockIdx.y * 32, c0 = blockIdx.x * 32;
#pragma unroll
    for (int i = 0; i < 4; ++i) {
        int r = ty + i * 8;
        tile[r][tx] = src[(long long)(r0 + r) * C + c0 + tx];
    }
    __syncthreads();
#pragma unroll
    for (int i = 0; i < 4; ++i) {
        int cc = ty + i * 8;  // becomes dst row
        dst[(long long)(c0 + cc) * R + r0 + tx] = (bf16)tile[tx][cc];
    }
}

// ---------------------------------------------------------------- generic GEMM
// C[M,N] (f32) = alpha * A[M,K](bf16) @ B[K,N] + bias[N] (+ C if accum)
// B is supplied PRE-TRANSPOSED: Bt[N,K] row-major (ldbt == K).
// 256 threads (8 waves); block tile 128x128, BK=32.
// Wave grid 4x2: each wave owns a 32x64 sub-tile (2 A-frags, 4 B-frags, 8 WMMA).
__global__ __launch_bounds__(256)
void gemm_bf16(const bf16* __restrict__ A, int lda,
               const bf16* __restrict__ Bt, int ldbt,
               float* __restrict__ C, int ldc,
               int M, int N, int K,
               const float* __restrict__ bias,
               float alpha, int accum)
{
    __shared__ bf16 As[128 * 32];   // [row][k]
    __shared__ bf16 Bst[128 * 32];  // [n][k]

    const int tid  = threadIdx.x;
    const int wave = tid >> 5, lane = tid & 31;
    const int lh = lane >> 4, ln = lane & 15;
    const int wr = wave >> 1;       // 0..3  (row group of 32)
    const int wc = wave & 1;        // 0..1  (col group of 64)
    const int m0 = blockIdx.y * 128;
    const int n0 = blockIdx.x * 128;

    f32x8 acc[2][4];
#pragma unroll
    for (int i = 0; i < 2; ++i)
#pragma unroll
        for (int t = 0; t < 4; ++t)
#pragma unroll
            for (int r = 0; r < 8; ++r) acc[i][t][r] = 0.f;

    const int ar = tid >> 1, ac = (tid & 1) * 16;   // A tile load coords
    const int bn = tid >> 1, bc = (tid & 1) * 16;   // Bt tile load coords

    for (int k0 = 0; k0 < K; k0 += 32) {
        __syncthreads();
        {   // A tile: 128 rows x 32 k
            const bf16* g = A + (long long)(m0 + ar) * lda + k0 + ac;
            uint4 p0 = *(const uint4*)g;
            uint4 p1 = *(const uint4*)(g + 8);
            *(uint4*)&As[ar * 32 + ac]     = p0;
            *(uint4*)&As[ar * 32 + ac + 8] = p1;
        }
        {   // Bt tile: 128 n x 32 k (contiguous both sides)
            const bf16* g = Bt + (long long)(n0 + bn) * ldbt + k0 + bc;
            uint4 p0 = *(const uint4*)g;
            uint4 p1 = *(const uint4*)(g + 8);
            *(uint4*)&Bst[bn * 32 + bc]     = p0;
            *(uint4*)&Bst[bn * 32 + bc + 8] = p1;
        }
        if (k0 + 32 < K) {  // prefetch next tiles (global_prefetch_b8)
            __builtin_prefetch(A + (long long)(m0 + ar) * lda + k0 + 32 + ac, 0, 1);
            __builtin_prefetch(Bt + (long long)(n0 + bn) * ldbt + k0 + 32 + bc, 0, 1);
        }
        __syncthreads();

        // gather all fragments first, then an uninterrupted WMMA chain
        bf16x16 fa[2], fb[4];
#pragma unroll
        for (int i = 0; i < 2; ++i)
            fa[i] = load_frag_a(&As[(wr * 32 + i * 16 + ln) * 32 + lh * 8]);
#pragma unroll
        for (int t = 0; t < 4; ++t)
            fb[t] = load_frag_b(&Bst[(wc * 64 + t * 16 + ln) * 32 + lh * 8]);
#pragma unroll
        for (int i = 0; i < 2; ++i)
#pragma unroll
            for (int t = 0; t < 4; ++t)
                acc[i][t] = __builtin_amdgcn_wmma_f32_16x16x32_bf16(
                    false, fa[i], false, fb[t], (short)0, acc[i][t], false, false);
    }

#pragma unroll
    for (int i = 0; i < 2; ++i)
#pragma unroll
        for (int t = 0; t < 4; ++t) {
            int col = n0 + wc * 64 + t * 16 + ln;
#pragma unroll
            for (int r = 0; r < 8; ++r) {
                int row = m0 + wr * 32 + i * 16 + r + lh * 8;
                float v = acc[i][t][r] * alpha;
                if (bias) v += bias[col];
                float* p = &C[(long long)row * ldc + col];
                if (accum) v += *p;
                *p = v;
            }
        }
}

// ---------------------------------------------------------------- RMSNorm + RoPE
// block = 32 threads handles one (row, head). Writes Qb/Kb [h][row][d] bf16,
// Vbt [h][d][row] bf16 (transposed for flash P@V B-fragments).
__global__ __launch_bounds__(32)
void norm_rope(const float* __restrict__ qf, const float* __restrict__ kf,
               const float* __restrict__ vf,
               const float* __restrict__ nrope, const float* __restrict__ crope,
               const float* __restrict__ qs, const float* __restrict__ ks,
               const float* __restrict__ vs,
               bf16* __restrict__ Qb, bf16* __restrict__ Kb, bf16* __restrict__ Vbt)
{
    const int row = blockIdx.x, h = blockIdx.y, lane = threadIdx.x;
    const float* fr = (row < L_NOISE) ? &nrope[(long long)row * HD]
                                      : &crope[(long long)(row - L_NOISE) * HD];
    const long long base = (long long)row * DIM + h * HD;
    const float c0 = cosf(fr[lane]),      s0 = sinf(fr[lane]);
    const float c1 = cosf(fr[lane + 32]), s1 = sinf(fr[lane + 32]);
    const float c2 = cosf(fr[lane + 64]), s2 = sinf(fr[lane + 64]);
    const float c3 = cosf(fr[lane + 96]), s3 = sinf(fr[lane + 96]);

    for (int which = 0; which < 3; ++which) {
        const float* src = which == 0 ? qf : (which == 1 ? kf : vf);
        const float* scl = which == 0 ? qs : (which == 1 ? ks : vs);
        float x0 = src[base + lane], x1 = src[base + lane + 32];
        float x2 = src[base + lane + 64], x3 = src[base + lane + 96];
        float ss = x0 * x0 + x1 * x1 + x2 * x2 + x3 * x3;
        for (int m = 1; m < 32; m <<= 1) ss += __shfl_xor(ss, m, 32);
        float inv = rsqrtf(ss * (1.0f / HD) + EPSV);
        x0 *= inv * scl[lane];      x1 *= inv * scl[lane + 32];
        x2 *= inv * scl[lane + 64]; x3 *= inv * scl[lane + 96];
        float y0 = x0, y1 = x1, y2 = x2, y3 = x3;
        if (which < 2) {  // RoPE: rot = [-x_hi, x_lo]
            y0 = x0 * c0 - x2 * s0;
            y1 = x1 * c1 - x3 * s1;
            y2 = x2 * c2 + x0 * s2;
            y3 = x3 * c3 + x1 * s3;
        }
        if (which == 0) {
            bf16* d = Qb + ((long long)h * LFULL + row) * HD;
            d[lane] = (bf16)y0; d[lane + 32] = (bf16)y1;
            d[lane + 64] = (bf16)y2; d[lane + 96] = (bf16)y3;
        } else if (which == 1) {
            bf16* d = Kb + ((long long)h * LFULL + row) * HD;
            d[lane] = (bf16)y0; d[lane + 32] = (bf16)y1;
            d[lane + 64] = (bf16)y2; d[lane + 96] = (bf16)y3;
        } else {
            bf16* d = Vbt + (long long)h * HD * LFULL + row;
            d[(long long)lane * LFULL]        = (bf16)y0;
            d[(long long)(lane + 32) * LFULL] = (bf16)y1;
            d[(long long)(lane + 64) * LFULL] = (bf16)y2;
            d[(long long)(lane + 96) * LFULL] = (bf16)y3;
        }
    }
}

// ---------------------------------------------------------------- Flash attention
// grid = (24 q-tiles, 16 heads), block = 256 (8 waves); wave owns 16 q-rows.
// Noise q-rows attend keys [0,3072); cond q-rows attend [2048,3072).
__global__ __launch_bounds__(256)
void flash_attn(const bf16* __restrict__ Qb, const bf16* __restrict__ Kb,
                const bf16* __restrict__ Vbt, float* __restrict__ xf)
{
    __shared__ bf16 Qs[128 * 128];   // [qrow][d]
    __shared__ bf16 Ks[32 * 128];    // [key][d]
    __shared__ bf16 Vst[128 * 32];   // [d][key]
    __shared__ bf16 Ps[8 * 16 * 32]; // per-wave P staging [m][key]

    const int h = blockIdx.y, q0 = blockIdx.x * 128;
    const int tid = threadIdx.x, wave = tid >> 5, lane = tid & 31;
    const int lh = lane >> 4, ln = lane & 15;
    const bf16* Qh = Qb + (long long)h * LFULL * HD;
    const bf16* Kh = Kb + (long long)h * LFULL * HD;
    const bf16* Vh = Vbt + (long long)h * HD * LFULL;

    // stage Q tile 128x128
    for (int i = 0; i < 4; ++i) {
        int idx = tid + i * 256;
        int r = idx >> 3, c = (idx & 7) * 16;
        const bf16* g = Qh + (long long)(q0 + r) * HD + c;
        *(uint4*)&Qs[r * 128 + c]     = *(const uint4*)g;
        *(uint4*)&Qs[r * 128 + c + 8] = *(const uint4*)(g + 8);
    }
    __syncthreads();

    bf16x16 qfrag[4];
#pragma unroll
    for (int f = 0; f < 4; ++f)
        qfrag[f] = load_frag_a(&Qs[(wave * 16 + ln) * 128 + f * 32 + lh * 8]);

    f32x8 o[8];
#pragma unroll
    for (int t = 0; t < 8; ++t)
#pragma unroll
        for (int r = 0; r < 8; ++r) o[t][r] = 0.f;
    float mst[8], lst[8];
#pragma unroll
    for (int r = 0; r < 8; ++r) { mst[r] = -1e30f; lst[r] = 0.f; }

    const float sc = 0.08838834764831845f;  // 1/sqrt(128)
    const int kstart = (q0 < L_NOISE) ? 0 : L_NOISE;
    const int kr = tid >> 3, kc = (tid & 7) * 16;   // K tile load coords
    const int vr = tid >> 1, vc = (tid & 1) * 16;   // V tile load coords

    for (int kb = kstart; kb < LFULL; kb += 32) {
        __syncthreads();
        {   // K tile 32x128
            const bf16* g = Kh + (long long)(kb + kr) * HD + kc;
            *(uint4*)&Ks[kr * 128 + kc]     = *(const uint4*)g;
            *(uint4*)&Ks[kr * 128 + kc + 8] = *(const uint4*)(g + 8);
        }
        {   // V tile [d][key] 128x32 (transposed in global already)
            const bf16* g = Vh + (long long)vr * LFULL + kb + vc;
            *(uint4*)&Vst[vr * 32 + vc]     = *(const uint4*)g;
            *(uint4*)&Vst[vr * 32 + vc + 8] = *(const uint4*)(g + 8);
        }
        if (kb + 32 < LFULL) {
            __builtin_prefetch(Kh + (long long)(kb + 32 + kr) * HD + kc, 0, 1);
            __builtin_prefetch(Vh + (long long)vr * LFULL + kb + 32 + vc, 0, 1);
        }
        __syncthreads();

        // S = Q @ K^T : two 16x16 tiles; gather 4 frags then 4 WMMA per tile
        f32x8 s[2];
#pragma unroll
        for (int t = 0; t < 2; ++t) {
            f32x8 a;
#pragma unroll
            for (int r = 0; r < 8; ++r) a[r] = 0.f;
            bf16x16 fb[4];
#pragma unroll
            for (int f = 0; f < 4; ++f)
                fb[f] = load_frag_b(&Ks[(t * 16 + ln) * 128 + f * 32 + lh * 8]);
#pragma unroll
            for (int f = 0; f < 4; ++f)
                a = __builtin_amdgcn_wmma_f32_16x16x32_bf16(
                    false, qfrag[f], false, fb[f], (short)0, a, false, false);
            s[t] = a;
        }

        // online softmax; VGPR r <-> q-row (r + 8*lh), column = lane within 16
        bf16* pbase = &Ps[wave * 512];
#pragma unroll
        for (int r = 0; r < 8; ++r) {
            float v = fmaxf(s[0][r], s[1][r]);
            for (int m = 1; m < 16; m <<= 1) v = fmaxf(v, __shfl_xor(v, m, 32));
            float rowmax = v * sc;
            float newm = fmaxf(mst[r], rowmax);
            float corr = __expf(mst[r] - newm);
            mst[r] = newm;
            float p0 = __expf(s[0][r] * sc - newm);
            float p1 = __expf(s[1][r] * sc - newm);
            float rs = p0 + p1;
            for (int m = 1; m < 16; m <<= 1) rs += __shfl_xor(rs, m, 32);
            lst[r] = lst[r] * corr + rs;
#pragma unroll
            for (int t = 0; t < 8; ++t) o[t][r] *= corr;
            pbase[(r + 8 * lh) * 32 + ln]      = (bf16)p0;
            pbase[(r + 8 * lh) * 32 + 16 + ln] = (bf16)p1;
        }
        __builtin_amdgcn_wave_barrier();  // DS ops are in-order per wave

        // P fragment (A-layout, K = 32 keys)
        bf16x16 pf = load_frag_a(&Ps[wave * 512 + ln * 32 + lh * 8]);

        // O += P @ V : gather 4 frags then 4 WMMA, twice
#pragma unroll
        for (int g2 = 0; g2 < 2; ++g2) {
            bf16x16 fv[4];
#pragma unroll
            for (int t = 0; t < 4; ++t)
                fv[t] = load_frag_b(&Vst[((g2 * 4 + t) * 16 + ln) * 32 + lh * 8]);
#pragma unroll
            for (int t = 0; t < 4; ++t)
                o[g2 * 4 + t] = __builtin_amdgcn_wmma_f32_16x16x32_bf16(
                    false, pf, false, fv[t], (short)0, o[g2 * 4 + t], false, false);
        }
    }

    float linv[8];
#pragma unroll
    for (int r = 0; r < 8; ++r) linv[r] = 1.0f / lst[r];
#pragma unroll
    for (int t = 0; t < 8; ++t) {
        int d = t * 16 + ln;
#pragma unroll
        for (int r = 0; r < 8; ++r) {
            int row = q0 + wave * 16 + r + 8 * lh;
            xf[(long long)row * DIM + h * HD + d] = o[t][r] * linv[r];
        }
    }
}

// ---------------------------------------------------------------- host
extern "C" void kernel_launch(void* const* d_in, const int* in_sizes, int n_in,
                              void* d_out, int out_size, void* d_ws, size_t ws_size,
                              hipStream_t stream)
{
    const float* noise = (const float*)d_in[0];
    const float* cond  = (const float*)d_in[1];
    const float* nrope = (const float*)d_in[2];
    const float* crope = (const float*)d_in[3];
    const float* Wq = (const float*)d_in[4];
    const float* Wk = (const float*)d_in[5];
    const float* Wv = (const float*)d_in[6];
    const float* Wo = (const float*)d_in[7];
    const float* bq = (const float*)d_in[8];
    const float* bk = (const float*)d_in[9];
    const float* bv = (const float*)d_in[10];
    const float* bo = (const float*)d_in[11];
    const float* qsc = (const float*)d_in[12];
    const float* ksc = (const float*)d_in[13];
    const float* vsc = (const float*)d_in[14];
    const float* lqd = (const float*)d_in[15];
    const float* lqu = (const float*)d_in[16];
    const float* lkd = (const float*)d_in[17];
    const float* lku = (const float*)d_in[18];
    const float* lvd = (const float*)d_in[19];
    const float* lvu = (const float*)d_in[20];
    const float* lod = (const float*)d_in[21];
    const float* lou = (const float*)d_in[22];

    char* ws = (char*)d_ws;
    size_t off = 0;
    auto carve = [&](size_t bytes) -> char* {
        char* p = ws + off;
        off += (bytes + 255) & ~(size_t)255;
        return p;
    };
    const size_t LD = (size_t)LFULL * DIM;
    bf16* jointb = (bf16*)carve(LD * 2);
    // all weights stored TRANSPOSED in bf16 (GEMM B-operand layout [N][K])
    bf16* WqbT = (bf16*)carve((size_t)DIM * DIM * 2);
    bf16* WkbT = (bf16*)carve((size_t)DIM * DIM * 2);
    bf16* WvbT = (bf16*)carve((size_t)DIM * DIM * 2);
    bf16* WobT = (bf16*)carve((size_t)DIM * DIM * 2);
    bf16* lqdT = (bf16*)carve((size_t)DIM * RANK_ * 2);  // [128][2048]
    bf16* lkdT = (bf16*)carve((size_t)DIM * RANK_ * 2);
    bf16* lvdT = (bf16*)carve((size_t)DIM * RANK_ * 2);
    bf16* lodT = (bf16*)carve((size_t)DIM * RANK_ * 2);
    bf16* lquT = (bf16*)carve((size_t)RANK_ * DIM * 2);  // [2048][128]
    bf16* lkuT = (bf16*)carve((size_t)RANK_ * DIM * 2);
    bf16* lvuT = (bf16*)carve((size_t)RANK_ * DIM * 2);
    bf16* louT = (bf16*)carve((size_t)RANK_ * DIM * 2);
    float* qf = (float*)carve(LD * 4);
    float* kf = (float*)carve(LD * 4);
    float* vf = (float*)carve(LD * 4);
    float* midf = (float*)carve((size_t)L_COND * RANK_ * 4);
    bf16*  midb = (bf16*)carve((size_t)L_COND * RANK_ * 2);
    bf16* Qb  = (bf16*)carve(LD * 2);
    bf16* Kb  = (bf16*)carve(LD * 2);
    bf16* Vbt = (bf16*)carve(LD * 2);
    float* xf = (float*)carve(LD * 4);
    bf16*  xb = (bf16*)carve(LD * 2);

    auto cast = [&](const float* s, bf16* d, long long n) {
        long long n2 = n >> 1;  // all our sizes are even
        int blocks = (int)((n2 + 1023) / 1024); if (blocks > 2048) blocks = 2048;
        if (blocks < 1) blocks = 1;
        cast_f32_bf16x2<<<dim3(blocks), dim3(256), 0, stream>>>(
            (const float2*)s, (unsigned*)d, n2);
    };
    auto castT = [&](const float* s, bf16* d, int R, int C) {
        cast_transpose_bf16<<<dim3(C / 32, R / 32), dim3(256), 0, stream>>>(s, d, R, C);
    };
    auto gemm = [&](const bf16* A, int lda, const bf16* Bt, int ldbt,
                    float* C, int ldc, int M, int N, int K,
                    const float* bias, float alpha, int accum) {
        gemm_bf16<<<dim3(N / 128, M / 128), dim3(256), 0, stream>>>(
            A, lda, Bt, ldbt, C, ldc, M, N, K, bias, alpha, accum);
    };

    // stage 0: casts (weights transposed)
    cast(noise, jointb, (long long)L_NOISE * DIM);
    cast(cond, jointb + (size_t)L_NOISE * DIM, (long long)L_COND * DIM);
    castT(Wq, WqbT, DIM, DIM);
    castT(Wk, WkbT, DIM, DIM);
    castT(Wv, WvbT, DIM, DIM);
    castT(Wo, WobT, DIM, DIM);
    castT(lqd, lqdT, DIM, RANK_);   // 2048x128 -> [128][2048]
    castT(lkd, lkdT, DIM, RANK_);
    castT(lvd, lvdT, DIM, RANK_);
    castT(lod, lodT, DIM, RANK_);
    castT(lqu, lquT, RANK_, DIM);   // 128x2048 -> [2048][128]
    castT(lku, lkuT, RANK_, DIM);
    castT(lvu, lvuT, RANK_, DIM);
    castT(lou, louT, RANK_, DIM);

    const bf16* jcond = jointb + (size_t)L_NOISE * DIM;
    const float scale_lora = 128.0f / 128.0f;  // ALPHA / RANK

    // stage 1: QKV projections (+bias) then LoRA on cond rows
    gemm(jointb, DIM, WqbT, DIM, qf, DIM, LFULL, DIM, DIM, bq, 1.0f, 0);
    gemm(jointb, DIM, WkbT, DIM, kf, DIM, LFULL, DIM, DIM, bk, 1.0f, 0);
    gemm(jointb, DIM, WvbT, DIM, vf, DIM, LFULL, DIM, DIM, bv, 1.0f, 0);

    float* tails[3] = { qf + (size_t)L_NOISE * DIM, kf + (size_t)L_NOISE * DIM,
                        vf + (size_t)L_NOISE * DIM };
    const bf16* downs[3] = { lqdT, lkdT, lvdT };
    const bf16* ups[3]   = { lquT, lkuT, lvuT };
    for (int i = 0; i < 3; ++i) {
        gemm(jcond, DIM, downs[i], DIM, midf, RANK_, L_COND, RANK_, DIM,
             nullptr, 1.0f, 0);
        cast(midf, midb, (long long)L_COND * RANK_);
        gemm(midb, RANK_, ups[i], RANK_, tails[i], DIM, L_COND, DIM, RANK_,
             nullptr, scale_lora, 1);
    }

    // stage 2: RMSNorm + RoPE -> per-head bf16 (V transposed)
    norm_rope<<<dim3(LFULL, NH), dim3(32), 0, stream>>>(
        qf, kf, vf, nrope, crope, qsc, ksc, vsc, Qb, Kb, Vbt);

    // stage 3: flash attention
    flash_attn<<<dim3(LFULL / 128, NH), dim3(256), 0, stream>>>(Qb, Kb, Vbt, xf);

    // stage 4: output projection directly into d_out, then LoRA-o on cond rows
    cast(xf, xb, (long long)LFULL * DIM);
    float* outp = (float*)d_out;
    gemm(xb, DIM, WobT, DIM, outp, DIM, LFULL, DIM, DIM, bo, 1.0f, 0);
    gemm(xb + (size_t)L_NOISE * DIM, DIM, lodT, DIM, midf, RANK_,
         L_COND, RANK_, DIM, nullptr, 1.0f, 0);
    cast(midf, midb, (long long)L_COND * RANK_);
    gemm(midb, RANK_, louT, RANK_, outp + (size_t)L_NOISE * DIM, DIM,
         L_COND, DIM, RANK_, nullptr, scale_lora, 1);
}
